// MultiHeadAttention_44143673868800
// MI455X (gfx1250) — matrix operand order repaired and checked by reading, not compiled, not run on hardware
//
#include <hip/hip_runtime.h>

// ---------------------------------------------------------------------------
// MHA for MI455X (gfx1250): bf16 WMMA everywhere, async-LDS staged GEMM B tiles.
// B=2, T=2048, C=1024, H=16, DQK=DV=64.
// ---------------------------------------------------------------------------

constexpr int Bb = 2, Tt = 2048, Cc = 1024, Hh = 16, Dd = 64;
constexpr int Mrows = Bb * Tt;        // 4096
constexpr int Nproj = Hh * Dd;        // 1024

typedef __attribute__((ext_vector_type(16))) __bf16 bf16x16;
typedef __attribute__((ext_vector_type(8)))  float  f32x8;

union FragAB {
    bf16x16 v;
    uint4   q[2];   // 32 bytes
};

// round-to-nearest-even fp32 -> bf16 (as raw u16)
static __device__ __forceinline__ unsigned short f2bf(float f) {
    unsigned int u = __float_as_uint(f);
    unsigned int r = (u + 0x7FFFu + ((u >> 16) & 1u)) >> 16;
    return (unsigned short)r;
}

// ---------------------------------------------------------------------------
// Async global->LDS copy (gfx1250 ASYNCcnt path) with synchronous fallback.
// Builtin signature (from compiler diagnostic): param0 = v4i* (global),
// param1 = v4i* (LDS), then imm offset, imm cpol.
// ---------------------------------------------------------------------------
#if defined(__AMDGCN__) && __has_builtin(__builtin_amdgcn_global_load_async_to_lds_b128)
#define HAVE_ASYNC_LDS 1
typedef int v4i_vec __attribute__((vector_size(16)));
typedef __attribute__((address_space(1))) v4i_vec gv4i;
typedef __attribute__((address_space(3))) v4i_vec lv4i;
static __device__ __forceinline__ void async_copy_b128(const void* g, void* l) {
    __builtin_amdgcn_global_load_async_to_lds_b128(
        (gv4i*)(unsigned long long)g,
        (lv4i*)(unsigned int)(unsigned long long)l,
        0, 0);
}
#if __has_builtin(__builtin_amdgcn_s_wait_asynccnt)
#define WAIT_ASYNC(n) __builtin_amdgcn_s_wait_asynccnt(n)
#else
#define WAIT_ASYNC(n) asm volatile("s_wait_asynccnt %0" ::"i"(n) : "memory")
#endif
#else
#define HAVE_ASYNC_LDS 0
static __device__ __forceinline__ void async_copy_b128(const void* g, void* l) {
    *(uint4*)l = *(const uint4*)g;   // global_load_b128 + ds_store_b128
}
#define WAIT_ASYNC(n) do {} while (0)
#endif

// ---------------------------------------------------------------------------
// fp32 -> bf16 elementwise cast (vectorized x4)
// ---------------------------------------------------------------------------
__global__ void cast_bf16_kernel(const float* __restrict__ in,
                                 unsigned short* __restrict__ out, int n4) {
    int i = blockIdx.x * blockDim.x + threadIdx.x;
    if (i >= n4) return;
    float4 f = ((const float4*)in)[i];
    unsigned int lo = (unsigned int)f2bf(f.x) | ((unsigned int)f2bf(f.y) << 16);
    unsigned int hi = (unsigned int)f2bf(f.z) | ((unsigned int)f2bf(f.w) << 16);
    ((uint2*)out)[i] = make_uint2(lo, hi);
}

// ---------------------------------------------------------------------------
// W (K x N) fp32 -> Wt (N x K) bf16, LDS-tiled transpose (16x16 tiles)
// ---------------------------------------------------------------------------
__global__ void transpose_cast_kernel(const float* __restrict__ W,
                                      unsigned short* __restrict__ Wt,
                                      int Kd, int Nd) {
    __shared__ float tile[16][17];
    int bx = blockIdx.x * 16;           // N
    int by = blockIdx.y * 16;           // K
    int tx = threadIdx.x & 15;
    int ty = threadIdx.x >> 4;
    tile[ty][tx] = W[(size_t)(by + ty) * Nd + (bx + tx)];
    __syncthreads();
    Wt[(size_t)(bx + ty) * Kd + (by + tx)] = f2bf(tile[tx][ty]);
}

// ---------------------------------------------------------------------------
// WMMA GEMM: D(M x N) = A(M x K) * Wt(N x K)^T + bias
// Block = 4 waves = 64x64 tile; wave = 16x64 strip (4 v8f accumulators).
// B tile (64n x 32k, shared by all 4 waves) is double-buffered in LDS and
// filled with async global->LDS copies (ASYNCcnt) each k-step.
// OutMode 0: bf16 head-split (B,H,T,Dd)             [q, k]
// OutMode 1: bf16 head-split transposed (B,H,Dd,T)  [v]
// OutMode 2: fp32 row-major                         [final projection]
// ---------------------------------------------------------------------------
template <int OutMode>
__global__ void gemm_wmma_kernel(const __bf16* __restrict__ A,
                                 const __bf16* __restrict__ Wt,
                                 const float* __restrict__ bias,
                                 void* __restrict__ Out,
                                 int M, int N, int K) {
    __shared__ __align__(16) unsigned short Bs[2][64 * 32];   // 2 x 4 KB

    const int tid  = threadIdx.x;
    const int wave = tid >> 5;
    const int lane = tid & 31;
    const int row  = lane & 15;
    const int kh   = lane >> 4;

    const int m0 = blockIdx.y * 64 + wave * 16;
    const int n0 = blockIdx.x * 64;

    const __bf16* Ap = A + (size_t)(m0 + row) * K;

    // stage one 64x32 B tile: 256 x 16B chunks, 2 per thread (2 async per wave)
    auto issueB = [&](int k0, int bufi) {
#pragma unroll
        for (int u = 0; u < 2; ++u) {
            const int c  = tid * 2 + u;
            const int n  = c >> 2;        // 0..63
            const int kc = c & 3;         // 0..3  (8 bf16 = 16B per chunk)
            async_copy_b128(Wt + (size_t)(n0 + n) * K + k0 + kc * 8,
                            &Bs[bufi][n * 32 + kc * 8]);
        }
    };

    issueB(0, 0);

    f32x8 acc[4] = {};
    int buf = 0;
    for (int k0 = 0; k0 < K; k0 += 32) {
        if (k0 + 32 < K) {
            issueB(k0 + 32, buf ^ 1);
            WAIT_ASYNC(2);                // current buffer's 2 loads are done
        } else {
            WAIT_ASYNC(0);
        }
        __syncthreads();                  // all waves' chunks visible

        // A fragment: lane row = m0+row; K chunks [k0+kh*8,+8) and [k0+16+kh*8,+8)
        FragAB a;
        a.q[0] = *(const uint4*)(Ap + k0 + kh * 8);
        a.q[1] = *(const uint4*)(Ap + k0 + 16 + kh * 8);
        __builtin_prefetch(Ap + k0 + 64, 0, 0);
#pragma unroll
        for (int j = 0; j < 4; ++j) {
            // B fragment from LDS: column n = j*16+row; K chunks [kh*8,+8) and [16+kh*8,+8)
            const unsigned short* br = &Bs[buf][(j * 16 + row) * 32 + kh * 8];
            FragAB b;
            b.q[0] = *(const uint4*)(br);
            b.q[1] = *(const uint4*)(br + 16);
            acc[j] = __builtin_amdgcn_wmma_f32_16x16x32_bf16(
                false, a.v, false, b.v, (short)0, acc[j], false, false);
        }
        __syncthreads();                  // done reading buf before it's refilled
        buf ^= 1;
    }

#pragma unroll
    for (int j = 0; j < 4; ++j) {
        const int n  = n0 + j * 16 + row;       // D: N = lane&15
        const float bv = bias[n];
#pragma unroll
        for (int r = 0; r < 8; ++r) {
            const int m = m0 + r + 8 * kh;      // D: M = r + 8*kh
            const float val = acc[j][r] + bv;
            if (OutMode == 2) {
                ((float*)Out)[(size_t)m * N + n] = val;
            } else {
                const unsigned short bw = f2bf(val);
                const int bb = m / Tt, t = m % Tt;
                const int h = n / Dd, dd = n % Dd;
                if (OutMode == 0) {
                    ((unsigned short*)Out)[(((size_t)(bb * Hh + h)) * Tt + t) * Dd + dd] = bw;
                } else {
                    ((unsigned short*)Out)[(((size_t)(bb * Hh + h)) * Dd + dd) * Tt + t] = bw;
                }
            }
        }
    }
}

// ---------------------------------------------------------------------------
// Flash attention: one wave per 16-row query tile, online softmax, causal.
// qh, kh: (B,H,T,Dd) bf16 ; vT: (B,H,Dd,T) bf16 ; out: (B,T,H*Dd) bf16
// ---------------------------------------------------------------------------
__global__ void attn_flash_kernel(const __bf16* __restrict__ qh,
                                  const __bf16* __restrict__ khd,
                                  const __bf16* __restrict__ vT,
                                  unsigned short* __restrict__ attn_out) {
    constexpr int WPB = 4;
    __shared__ __align__(16) unsigned short pLds[WPB][16 * 32];

    const int wave = threadIdx.x >> 5;
    const int lane = threadIdx.x & 31;
    const int row  = lane & 15;
    const int kh   = lane >> 4;

    const int qt = blockIdx.x * WPB + wave;   // 0..127
    const int bh = blockIdx.y;                // 0..B*H-1
    const int q0 = qt * 16;

    // Q fragments for k-dims [0,32) and [32,64): stay resident in VGPRs
    const __bf16* qp = qh + ((size_t)bh * Tt + q0 + row) * Dd;
    FragAB aq0, aq1;
    aq0.q[0] = *(const uint4*)(qp + kh * 8);
    aq0.q[1] = *(const uint4*)(qp + 16 + kh * 8);
    aq1.q[0] = *(const uint4*)(qp + 32 + kh * 8);
    aq1.q[1] = *(const uint4*)(qp + 48 + kh * 8);

    float mrow[8], lrow[8];
#pragma unroll
    for (int r = 0; r < 8; ++r) { mrow[r] = -1e30f; lrow[r] = 0.f; }
    f32x8 oacc[4] = {};

    const float scale = 0.125f;               // 1/sqrt(64)
    const int kmax = q0 + 16;                 // causal bound (exclusive)

    for (int ks = 0; ks < kmax; ks += 32) {
        // ---- scores for 32 keys: two 16-key sub-blocks, K-dim 64 in 2 steps
        f32x8 s[2];
#pragma unroll
        for (int sb = 0; sb < 2; ++sb) {
            const int kb = ks + sb * 16;
            const __bf16* kp = khd + ((size_t)bh * Tt + kb + row) * Dd;
            FragAB b0, b1;
            b0.q[0] = *(const uint4*)(kp + kh * 8);
            b0.q[1] = *(const uint4*)(kp + 16 + kh * 8);
            b1.q[0] = *(const uint4*)(kp + 32 + kh * 8);
            b1.q[1] = *(const uint4*)(kp + 48 + kh * 8);
            f32x8 sc = {};
            sc = __builtin_amdgcn_wmma_f32_16x16x32_bf16(
                false, aq0.v, false, b0.v, (short)0, sc, false, false);
            sc = __builtin_amdgcn_wmma_f32_16x16x32_bf16(
                false, aq1.v, false, b1.v, (short)0, sc, false, false);
            s[sb] = sc;
        }
        // ---- scale + causal mask (element (r): qi = q0+r+8*kh, kj = kb+row)
#pragma unroll
        for (int sb = 0; sb < 2; ++sb) {
            const int kj = ks + sb * 16 + row;
#pragma unroll
            for (int r = 0; r < 8; ++r) {
                const int qi = q0 + r + 8 * kh;
                const float v = s[sb][r] * scale;
                s[sb][r] = (kj <= qi) ? v : -1e30f;
            }
        }
        // ---- online softmax update (row = 16 lanes of one half-wave)
#pragma unroll
        for (int r = 0; r < 8; ++r) {
            float v = fmaxf(s[0][r], s[1][r]);
#pragma unroll
            for (int off = 8; off; off >>= 1)
                v = fmaxf(v, __shfl_xor(v, off, 16));
            const float mn = fmaxf(mrow[r], v);
            const float p0 = __expf(s[0][r] - mn);
            const float p1 = __expf(s[1][r] - mn);
            s[0][r] = p0; s[1][r] = p1;
            float rs = p0 + p1;
#pragma unroll
            for (int off = 8; off; off >>= 1)
                rs += __shfl_xor(rs, off, 16);
            const float corr = __expf(mrow[r] - mn);
            lrow[r] = lrow[r] * corr + rs;
            mrow[r] = mn;
#pragma unroll
            for (int j = 0; j < 4; ++j) oacc[j][r] *= corr;
        }
        // ---- P: D-layout -> A-layout via per-wave LDS staging (16x32 bf16)
        unsigned short* pw = pLds[wave];
#pragma unroll
        for (int sb = 0; sb < 2; ++sb)
#pragma unroll
            for (int r = 0; r < 8; ++r)
                pw[(r + 8 * kh) * 32 + sb * 16 + row] = f2bf(s[sb][r]);
        FragAB pa;
        {
            const unsigned short* pr = pw + row * 32;
            pa.q[0] = *(const uint4*)(pr + kh * 8);
            pa.q[1] = *(const uint4*)(pr + 16 + kh * 8);
        }
        // ---- O += P @ V  (B from vT: column dv = j*16+row, keys contiguous)
#pragma unroll
        for (int j = 0; j < 4; ++j) {
            const __bf16* vp = vT + ((size_t)bh * Dd + j * 16 + row) * Tt + ks;
            FragAB b;
            b.q[0] = *(const uint4*)(vp + kh * 8);
            b.q[1] = *(const uint4*)(vp + 16 + kh * 8);
            oacc[j] = __builtin_amdgcn_wmma_f32_16x16x32_bf16(
                false, pa.v, false, b.v, (short)0, oacc[j], false, false);
        }
    }

    // ---- normalize + write (B,T,H*Dd) bf16
    const int bb = bh / Hh, h = bh % Hh;
#pragma unroll
    for (int j = 0; j < 4; ++j) {
        const int dd = j * 16 + row;
#pragma unroll
        for (int r = 0; r < 8; ++r) {
            const int t = q0 + r + 8 * kh;
            const float val = oacc[j][r] / lrow[r];
            attn_out[((size_t)(bb * Tt + t)) * (Hh * Dd) + h * Dd + dd] = f2bf(val);
        }
    }
}

// ---------------------------------------------------------------------------
// launch
// ---------------------------------------------------------------------------
extern "C" void kernel_launch(void* const* d_in, const int* in_sizes, int n_in,
                              void* d_out, int out_size, void* d_ws, size_t ws_size,
                              hipStream_t stream) {
    (void)in_sizes; (void)n_in; (void)out_size; (void)ws_size;

    const float* Q  = (const float*)d_in[0];
    const float* K  = (const float*)d_in[1];
    const float* V  = (const float*)d_in[2];
    /* d_in[3] = causal mask: computed analytically, unused */
    const float* Wq = (const float*)d_in[4];
    const float* bq = (const float*)d_in[5];
    const float* Wk = (const float*)d_in[6];
    const float* bk = (const float*)d_in[7];
    const float* Wv = (const float*)d_in[8];
    const float* bv = (const float*)d_in[9];
    const float* Wo = (const float*)d_in[10];
    const float* bo = (const float*)d_in[11];

    // workspace carve-up (total 64 MB)
    constexpr size_t SZ_ACT = (size_t)Mrows * Cc * 2;       // 8 MB
    constexpr size_t SZ_W   = (size_t)Cc * Nproj * 2;       // 2 MB
    char* p = (char*)d_ws;
    unsigned short* Qbf  = (unsigned short*)p; p += SZ_ACT;
    unsigned short* Kbf  = (unsigned short*)p; p += SZ_ACT;
    unsigned short* Vbf  = (unsigned short*)p; p += SZ_ACT;
    unsigned short* WqT  = (unsigned short*)p; p += SZ_W;
    unsigned short* WkT  = (unsigned short*)p; p += SZ_W;
    unsigned short* WvT  = (unsigned short*)p; p += SZ_W;
    unsigned short* WoT  = (unsigned short*)p; p += SZ_W;
    unsigned short* qh   = (unsigned short*)p; p += SZ_ACT;
    unsigned short* khd  = (unsigned short*)p; p += SZ_ACT;
    unsigned short* vT   = (unsigned short*)p; p += SZ_ACT;
    unsigned short* attn = (unsigned short*)p; p += SZ_ACT;

    // 1) casts
    const int nAct4 = Mrows * Cc / 4;                       // 1,048,576
    cast_bf16_kernel<<<nAct4 / 256, 256, 0, stream>>>(Q, Qbf, nAct4);
    cast_bf16_kernel<<<nAct4 / 256, 256, 0, stream>>>(K, Kbf, nAct4);
    cast_bf16_kernel<<<nAct4 / 256, 256, 0, stream>>>(V, Vbf, nAct4);

    dim3 tgrid(Nproj / 16, Cc / 16);
    transpose_cast_kernel<<<tgrid, 256, 0, stream>>>(Wq, WqT, Cc, Nproj);
    transpose_cast_kernel<<<tgrid, 256, 0, stream>>>(Wk, WkT, Cc, Nproj);
    transpose_cast_kernel<<<tgrid, 256, 0, stream>>>(Wv, WvT, Cc, Nproj);
    transpose_cast_kernel<<<tgrid, 256, 0, stream>>>(Wo, WoT, Nproj, Cc);

    // 2) QKV projections (head-split; V transposed within head)
    dim3 ggrid(Nproj / 64, Mrows / 64);                     // (16, 64)
    gemm_wmma_kernel<0><<<ggrid, 128, 0, stream>>>(
        (const __bf16*)Qbf, (const __bf16*)WqT, bq, qh,  Mrows, Nproj, Cc);
    gemm_wmma_kernel<0><<<ggrid, 128, 0, stream>>>(
        (const __bf16*)Kbf, (const __bf16*)WkT, bk, khd, Mrows, Nproj, Cc);
    gemm_wmma_kernel<1><<<ggrid, 128, 0, stream>>>(
        (const __bf16*)Vbf, (const __bf16*)WvT, bv, vT,  Mrows, Nproj, Cc);

    // 3) causal flash attention: grid.x * 4 waves = 128 query tiles
    dim3 agrid((Tt / 16) / 4, Bb * Hh);                     // (32, 32)
    attn_flash_kernel<<<agrid, 128, 0, stream>>>(
        (const __bf16*)qh, (const __bf16*)khd, (const __bf16*)vT, attn);

    // 4) output projection -> fp32
    gemm_wmma_kernel<2><<<ggrid, 128, 0, stream>>>(
        (const __bf16*)attn, (const __bf16*)WoT, bo, d_out, Mrows, Cc, Nproj);
}
